// MultiHeadedAttention_55851754717545
// MI455X (gfx1250) — compile-verified, matrix operand order checked by reading, not run on hardware
//
#include <hip/hip_runtime.h>

// ---------------------------------------------------------------------------
// MultiHeadedAttention on MI455X (gfx1250, wave32, WMMA + async-LDS/TDM)
//   B=2, S=2048, D_MODEL=1024, H=16, D_K=64, MAX_REL=32
// ---------------------------------------------------------------------------

typedef __attribute__((ext_vector_type(16))) _Float16 v16h;
typedef __attribute__((ext_vector_type(8)))  _Float16 v8h;
typedef __attribute__((ext_vector_type(8)))  float    v8f;
typedef __attribute__((ext_vector_type(4)))  unsigned v4u;
typedef __attribute__((ext_vector_type(8)))  int      v8i;
typedef __attribute__((ext_vector_type(4)))  int      v4i;

#define MHA_B      2
#define MHA_S      2048
#define MHA_D      1024
#define MHA_H      16
#define MHA_DK     64
#define MHA_MAXREL 32

#ifndef __has_builtin
#define __has_builtin(x) 0
#endif
#if __has_builtin(__builtin_amdgcn_tensor_load_to_lds)
#define HAVE_TDM 1
#else
#define HAVE_TDM 0
#endif

// s_wait_asynccnt <= n  (async-to-LDS completion, in-order per wave)
#define S_WAIT_ASYNC(n) asm volatile("s_wait_asynccnt " #n ::: "memory")

// One 16B async global->LDS copy (GLOBAL_LOAD_ASYNC_TO_LDS_B128, ASYNCcnt).
// Generic LDS address truncates to the LDS byte offset (ISA 10.2 aperture).
__device__ __forceinline__ void mha_async_b128(void* lds, const void* g) {
    unsigned l = (unsigned)(uintptr_t)lds;
    asm volatile("global_load_async_to_lds_b128 %0, %1, off"
                 :: "v"(l), "v"((unsigned long long)(uintptr_t)g)
                 : "memory");
}

#if HAVE_TDM
// TDM: DMA a 2D f16 tile [tile_h rows x 32 halves] into LDS with a 40-half
// (80B) pitch. D# per CDNA5 ISA: pad_interval=3 (16 DWORDs = one 32-half
// row), pad_amount=3 (4 DWORDs = 8 halves) -> LDS pitch 40 halves.
__device__ __forceinline__ void mha_tdm_tile32(const _Float16* gptr, void* ldsptr,
                                               int tile_h, int row_stride_elems) {
    unsigned long long ga = (unsigned long long)(uintptr_t)gptr;
    v4u g0 = { 1u,                                   // count=1, user mode
               (unsigned)(uintptr_t)ldsptr,          // lds_addr
               (unsigned)ga,                         // global_addr[31:0]
               (unsigned)((ga >> 32) & 0x01FFFFFFu) | (2u << 30) }; // type=2
    const unsigned td0 = (unsigned)row_stride_elems; // tensor_dim0
    const unsigned td1 = 1u << 20;                   // tensor_dim1 (large)
    const unsigned d0 = (1u << 16)    // data_size = 2 bytes
                      | (1u << 20)    // pad_enable
                      | (3u << 22)    // pad_interval: 16 DWORDs
                      | (3u << 25);   // pad_amount:   4 DWORDs
    v8i g1 = { (int)d0,
               (int)((td0 & 0xFFFFu) << 16),
               (int)((td0 >> 16) | ((td1 & 0xFFFFu) << 16)),
               (int)((td1 >> 16) | (32u << 16)),            // tile_dim0 = 32
               (int)(unsigned)tile_h,                       // tile_dim1
               (int)(unsigned)row_stride_elems,             // dim0_stride lo
               0, 0 };
    v4i z4 = {0, 0, 0, 0};
    v8i z8 = {0, 0, 0, 0, 0, 0, 0, 0};
    // 6-arg form (amdgpu-toolchain / clang-23): groups 2,3 unused for 2D.
    __builtin_amdgcn_tensor_load_to_lds(g0, g1, z4, z4, z8, 0);
}
#endif

__device__ __forceinline__ v16h mha_ldfrag(const _Float16* p) {
    v16h a;
    *reinterpret_cast<v8h*>(&a)       = *reinterpret_cast<const v8h*>(p);
    *(reinterpret_cast<v8h*>(&a) + 1) = *reinterpret_cast<const v8h*>(p + 16);
    return a;
}

__device__ __forceinline__ v8f mha_wmma(v16h a, v16h b, v8f c) {
    return __builtin_amdgcn_wmma_f32_16x16x32_f16(
        false, a, false, b, (short)0, c, false, false);
}

// ---------------------------------------------------------------------------
// Elementwise f32 -> f16 conversion. n multiple of 4.
// ---------------------------------------------------------------------------
__global__ void mha_cvt_f16(const float* __restrict__ in,
                            _Float16* __restrict__ out, int n) {
    int i = (blockIdx.x * blockDim.x + threadIdx.x) * 4;
    if (i < n) {
        float4 v = *reinterpret_cast<const float4*>(in + i);
        out[i + 0] = (_Float16)v.x; out[i + 1] = (_Float16)v.y;
        out[i + 2] = (_Float16)v.z; out[i + 3] = (_Float16)v.w;
    }
}

__global__ void mha_rel_reduce(const float* __restrict__ rel_emb,
                               float* __restrict__ rel_sums) {
    int r = threadIdx.x;
    if (r < 2 * MHA_MAXREL + 1) {
        float s = 0.f;
        #pragma unroll
        for (int d = 0; d < MHA_DK; ++d) s += rel_emb[r * MHA_DK + d];
        rel_sums[r] = s;
    }
}

// ---------------------------------------------------------------------------
// GEMM: C[M x 1024] = X[M x 1024] * W^T + bias   (f16 in, f32 acc)
// Block = 4 waves = 64x64 tile. Per 32-K step: A tile (64x32) via async-LDS,
// B tile (64x32, shared by all 4 waves) via TDM; double-buffered.
// mode 0: Qp[b,h,s,64]  1: Kp[b,h,s,64]  2: Vt[b,h,64,S]  3: f32 [M,1024]
// ---------------------------------------------------------------------------
__global__ __launch_bounds__(128)
void mha_gemm_wmma(const _Float16* __restrict__ X,
                   const _Float16* __restrict__ W,
                   const float* __restrict__ bias,
                   void* __restrict__ out, int mode) {
    __shared__ _Float16 Abuf[2][64][40];   // 80B pitch -> aligned ds_load_b128
    __shared__ _Float16 Bbuf[2][64][40];

    const int tid  = threadIdx.x;
    const int lane = tid & 31;
    const int wave = tid >> 5;
    const int lr   = lane & 15;
    const int hl   = lane >> 4;
    const int rowbase0 = blockIdx.x * 64;
    const int colbase  = blockIdx.y * 64;
    const int rowbase  = rowbase0 + wave * 16;

    auto stageA = [&](int k, int nb) {
        #pragma unroll
        for (int j = 0; j < 2; ++j) {
            int c = tid + j * 128;              // 256 chunks: 64 rows x 4 segs
            int row = c >> 2, seg = c & 3;
            mha_async_b128(&Abuf[nb][row][seg * 8],
                           X + (size_t)(rowbase0 + row) * MHA_D + k + seg * 8);
        }
    };
    auto stageB = [&](int k, int nb) {
#if HAVE_TDM
        if (wave == 0)
            mha_tdm_tile32(W + (size_t)colbase * MHA_D + k,
                           &Bbuf[nb][0][0], 64, MHA_D);
#else
        #pragma unroll
        for (int j = 0; j < 2; ++j) {
            int c = tid + j * 128;
            int row = c >> 2, seg = c & 3;
            mha_async_b128(&Bbuf[nb][row][seg * 8],
                           W + (size_t)(colbase + row) * MHA_D + k + seg * 8);
        }
#endif
    };

    const v8f zero = {};
    v8f acc[4] = {zero, zero, zero, zero};

    stageA(0, 0);
    stageB(0, 0);

    const int NSTEP = MHA_D / 32;               // 32
    for (int i = 0; i < NSTEP; ++i) {
        const int cur = i & 1;
        if (i + 1 < NSTEP) {
            stageA((i + 1) * 32, cur ^ 1);
            stageB((i + 1) * 32, cur ^ 1);
#if HAVE_TDM
            S_WAIT_ASYNC(2);
            if (wave == 0) __builtin_amdgcn_s_wait_tensorcnt(1);
#else
            S_WAIT_ASYNC(4);
#endif
        } else {
            S_WAIT_ASYNC(0);
#if HAVE_TDM
            if (wave == 0) __builtin_amdgcn_s_wait_tensorcnt(0);
#endif
        }
        __syncthreads();                        // tiles[cur] visible to block

        v16h a = mha_ldfrag(&Abuf[cur][wave * 16 + lr][hl * 8]);
        #pragma unroll
        for (int nt = 0; nt < 4; ++nt) {
            v16h b = mha_ldfrag(&Bbuf[cur][nt * 16 + lr][hl * 8]);
            acc[nt] = mha_wmma(a, b, acc[nt]);
        }
        __syncthreads();                        // reads done before overwrite
    }

    // Epilogue. C/D layout: elem r of lane l -> row r+8*(l>>4), col l&15.
    #pragma unroll
    for (int nt = 0; nt < 4; ++nt) {
        #pragma unroll
        for (int r = 0; r < 8; ++r) {
            const int col  = colbase + nt * 16 + lr;
            const int rowg = rowbase + r + 8 * hl;
            float val = acc[nt][r] + bias[col];
            if (mode == 3) {
                ((float*)out)[(size_t)rowg * MHA_D + col] = val;
            } else {
                const int bb = rowg >> 11, ss = rowg & (MHA_S - 1);
                const int hh = col >> 6,  dd = col & (MHA_DK - 1);
                _Float16* oh = (_Float16*)out;
                if (mode == 2)
                    oh[(((size_t)bb * MHA_H + hh) * MHA_DK + dd) * MHA_S + ss] =
                        (_Float16)val;
                else
                    oh[(((size_t)bb * MHA_H + hh) * MHA_S + ss) * MHA_DK + dd] =
                        (_Float16)val;
            }
        }
    }
}

// ---------------------------------------------------------------------------
// Flash-style attention. Grid (S/64, H, B), 4 waves/block (one 16-row q tile
// each, same (b,h)). K/V tiles for each 32-key step are shared by the whole
// block: staged via async-to-LDS, double-buffered. Online softmax with
// shfl_xor row reductions; P transposed through LDS; 8 WMMAs per step/wave.
// ---------------------------------------------------------------------------
__global__ __launch_bounds__(128)
void mha_attn_wmma(const _Float16* __restrict__ Qp,
                   const _Float16* __restrict__ Kp,
                   const _Float16* __restrict__ Vt,
                   const int* __restrict__ mask,
                   const float* __restrict__ rel_sums,
                   _Float16* __restrict__ attn16) {
    __shared__ _Float16 Kbuf[2][32][72];   // 144B pitch (16B aligned chunks)
    __shared__ _Float16 Vbuf[2][64][40];   // 80B pitch
    __shared__ _Float16 Pbuf[4][16][40];
    __shared__ float    relS[2 * MHA_MAXREL + 1];

    const int tid = threadIdx.x;
    if (tid < 2 * MHA_MAXREL + 1) relS[tid] = rel_sums[tid];

    const int lane = tid & 31;
    const int wave = tid >> 5;
    const int lr   = lane & 15;
    const int hl   = lane >> 4;
    const int b  = blockIdx.z, h = blockIdx.y;
    const int bh = b * MHA_H + h;
    const int qbase = blockIdx.x * 64 + wave * 16;

    const _Float16* Kbh = Kp + (size_t)bh * MHA_S * MHA_DK;
    const _Float16* Vbh = Vt + (size_t)bh * MHA_DK * MHA_S;

    auto stageKV = [&](int kb, int nb) {
        #pragma unroll
        for (int j = 0; j < 2; ++j) {      // K: 32 rows x 8 segs = 256 chunks
            int c = tid + j * 128;
            int row = c >> 3, seg = c & 7;
            mha_async_b128(&Kbuf[nb][row][seg * 8],
                           Kbh + (size_t)(kb + row) * MHA_DK + seg * 8);
        }
        #pragma unroll
        for (int j = 0; j < 2; ++j) {      // V: 64 d-rows x 4 segs = 256
            int c = tid + j * 128;
            int row = c >> 2, seg = c & 3;
            mha_async_b128(&Vbuf[nb][row][seg * 8],
                           Vbh + (size_t)row * MHA_S + kb + seg * 8);
        }
    };

    // Q fragments (K-dim 0..63), loaded once from global.
    const _Float16* qrowp = Qp + ((size_t)bh * MHA_S + qbase + lr) * MHA_DK;
    v16h qa0 = mha_ldfrag(qrowp + 0  + hl * 8);
    v16h qa1 = mha_ldfrag(qrowp + 32 + hl * 8);

    const v8f zero = {};
    v8f acc[4] = {zero, zero, zero, zero};
    float mrun[8], srun[8];
    #pragma unroll
    for (int r = 0; r < 8; ++r) { mrun[r] = -INFINITY; srun[r] = 0.f; }

    const int* mbase = mask + (size_t)b * MHA_S * MHA_S;

    stageKV(0, 0);

    const int NSTEP = MHA_S / 32;          // 64
    for (int i = 0; i < NSTEP; ++i) {
        const int kb  = i * 32;
        const int cur = i & 1;
        if (i + 1 < NSTEP) { stageKV(kb + 32, cur ^ 1); S_WAIT_ASYNC(4); }
        else               { S_WAIT_ASYNC(0); }
        __syncthreads();                   // K/V[cur] ready for whole block

        // ---- scores 16x32: 4 WMMAs from LDS ------------------------------
        v8f sc0 = zero, sc1 = zero;
        sc0 = mha_wmma(qa0, mha_ldfrag(&Kbuf[cur][lr]     [0  + hl * 8]), sc0);
        sc0 = mha_wmma(qa1, mha_ldfrag(&Kbuf[cur][lr]     [32 + hl * 8]), sc0);
        sc1 = mha_wmma(qa0, mha_ldfrag(&Kbuf[cur][16 + lr][0  + hl * 8]), sc1);
        sc1 = mha_wmma(qa1, mha_ldfrag(&Kbuf[cur][16 + lr][32 + hl * 8]), sc1);

        // ---- scale + rel bias + mask -------------------------------------
        float f0[8], f1[8];
        #pragma unroll
        for (int r = 0; r < 8; ++r) {
            const int qrow = qbase + r + 8 * hl;
            {
                const int kc = kb + lr;
                int rel = qrow - kc;
                rel = rel < -MHA_MAXREL ? -MHA_MAXREL
                    : (rel > MHA_MAXREL ? MHA_MAXREL : rel);
                float v = sc0[r] * 0.125f + relS[rel + MHA_MAXREL];
                f0[r] = mbase[(size_t)qrow * MHA_S + kc] ? v : -1e9f;
            }
            {
                const int kc = kb + 16 + lr;
                int rel = qrow - kc;
                rel = rel < -MHA_MAXREL ? -MHA_MAXREL
                    : (rel > MHA_MAXREL ? MHA_MAXREL : rel);
                float v = sc1[r] * 0.125f + relS[rel + MHA_MAXREL];
                f1[r] = mbase[(size_t)qrow * MHA_S + kc] ? v : -1e9f;
            }
        }

        // ---- online softmax: row = 16 lanes of one half-wave -------------
        #pragma unroll
        for (int r = 0; r < 8; ++r) {
            float t = fmaxf(f0[r], f1[r]);
            #pragma unroll
            for (int off = 1; off < 16; off <<= 1)
                t = fmaxf(t, __shfl_xor(t, off, 16));
            const float mn = fmaxf(mrun[r], t);
            const float p0 = __expf(f0[r] - mn);
            const float p1 = __expf(f1[r] - mn);
            float rs = p0 + p1;
            #pragma unroll
            for (int off = 1; off < 16; off <<= 1)
                rs += __shfl_xor(rs, off, 16);
            const float alpha = __expf(mrun[r] - mn);
            srun[r] = srun[r] * alpha + rs;
            mrun[r] = mn;
            #pragma unroll
            for (int nt = 0; nt < 4; ++nt) acc[nt][r] *= alpha;
            Pbuf[wave][r + 8 * hl][lr]      = (_Float16)p0;
            Pbuf[wave][r + 8 * hl][16 + lr] = (_Float16)p1;
        }
        __syncthreads();

        // ---- P (16x32) x V (32x64): 4 WMMAs from LDS ---------------------
        v16h pf = mha_ldfrag(&Pbuf[wave][lr][hl * 8]);
        #pragma unroll
        for (int nt = 0; nt < 4; ++nt) {
            v16h vf = mha_ldfrag(&Vbuf[cur][nt * 16 + lr][hl * 8]);
            acc[nt] = mha_wmma(pf, vf, acc[nt]);
        }
        __syncthreads();                   // buf[cur] reads done
    }

    // ---- normalize + write concat-head layout [b,s, h*64+d] --------------
    #pragma unroll
    for (int nt = 0; nt < 4; ++nt) {
        #pragma unroll
        for (int r = 0; r < 8; ++r) {
            const int qrow = qbase + r + 8 * hl;
            const int d    = nt * 16 + lr;
            float o = acc[nt][r] / srun[r];
            attn16[((size_t)b * MHA_S + qrow) * MHA_D + h * MHA_DK + d] =
                (_Float16)o;
        }
    }
}

// ---------------------------------------------------------------------------
// Host-side launch
// ---------------------------------------------------------------------------
extern "C" void kernel_launch(void* const* d_in, const int* in_sizes, int n_in,
                              void* d_out, int out_size, void* d_ws, size_t ws_size,
                              hipStream_t stream) {
    (void)in_sizes; (void)n_in; (void)out_size; (void)ws_size;

    const float* query   = (const float*)d_in[0];
    const float* key     = (const float*)d_in[1];
    const float* value   = (const float*)d_in[2];
    const int*   mask    = (const int*)  d_in[3];
    const float* Wq      = (const float*)d_in[4];
    const float* bq      = (const float*)d_in[5];
    const float* Wk      = (const float*)d_in[6];
    const float* bk      = (const float*)d_in[7];
    const float* Wv      = (const float*)d_in[8];
    const float* bv      = (const float*)d_in[9];
    const float* Wo      = (const float*)d_in[10];
    const float* bo      = (const float*)d_in[11];
    const float* rel_emb = (const float*)d_in[12];

    const size_t NX = (size_t)MHA_B * MHA_S * MHA_D;   // 4194304
    const size_t NW = (size_t)MHA_D * MHA_D;           // 1048576

    char* w = (char*)d_ws;
    _Float16* q16  = (_Float16*)w; w += NX * 2;
    _Float16* k16  = (_Float16*)w; w += NX * 2;
    _Float16* v16  = (_Float16*)w; w += NX * 2;
    _Float16* wq16 = (_Float16*)w; w += NW * 2;
    _Float16* wk16 = (_Float16*)w; w += NW * 2;
    _Float16* wv16 = (_Float16*)w; w += NW * 2;
    _Float16* wo16 = (_Float16*)w; w += NW * 2;
    _Float16* Qp   = (_Float16*)w; w += NX * 2;
    _Float16* Kp   = (_Float16*)w; w += NX * 2;
    _Float16* Vt   = (_Float16*)w; w += NX * 2;
    float*    rels = (float*)w;    w += 256;
    _Float16* attn16 = q16;   // reuse: q16 dead after Q projection

    const int cvtB = 256;
    hipLaunchKernelGGL(mha_cvt_f16, dim3((unsigned)(NX / 4 / cvtB)), dim3(cvtB),
                       0, stream, query, q16, (int)NX);
    hipLaunchKernelGGL(mha_cvt_f16, dim3((unsigned)(NX / 4 / cvtB)), dim3(cvtB),
                       0, stream, key, k16, (int)NX);
    hipLaunchKernelGGL(mha_cvt_f16, dim3((unsigned)(NX / 4 / cvtB)), dim3(cvtB),
                       0, stream, value, v16, (int)NX);
    hipLaunchKernelGGL(mha_cvt_f16, dim3((unsigned)(NW / 4 / cvtB)), dim3(cvtB),
                       0, stream, Wq, wq16, (int)NW);
    hipLaunchKernelGGL(mha_cvt_f16, dim3((unsigned)(NW / 4 / cvtB)), dim3(cvtB),
                       0, stream, Wk, wk16, (int)NW);
    hipLaunchKernelGGL(mha_cvt_f16, dim3((unsigned)(NW / 4 / cvtB)), dim3(cvtB),
                       0, stream, Wv, wv16, (int)NW);
    hipLaunchKernelGGL(mha_cvt_f16, dim3((unsigned)(NW / 4 / cvtB)), dim3(cvtB),
                       0, stream, Wo, wo16, (int)NW);
    hipLaunchKernelGGL(mha_rel_reduce, dim3(1), dim3(128), 0, stream,
                       rel_emb, rels);

    dim3 ggrid(4096 / 64, MHA_D / 64), gblk(128);
    hipLaunchKernelGGL(mha_gemm_wmma, ggrid, gblk, 0, stream,
                       q16, wq16, bq, (void*)Qp, 0);
    hipLaunchKernelGGL(mha_gemm_wmma, ggrid, gblk, 0, stream,
                       k16, wk16, bk, (void*)Kp, 1);
    hipLaunchKernelGGL(mha_gemm_wmma, ggrid, gblk, 0, stream,
                       v16, wv16, bv, (void*)Vt, 2);

    hipLaunchKernelGGL(mha_attn_wmma, dim3(MHA_S / 64, MHA_H, MHA_B), dim3(128),
                       0, stream, Qp, Kp, Vt, mask, rels, attn16);

    hipLaunchKernelGGL(mha_gemm_wmma, ggrid, gblk, 0, stream,
                       attn16, wo16, bo, d_out, 3);
}
